// Conv_KNRM_34084860461087
// MI455X (gfx1250) — matrix operand order repaired and checked by reading, not compiled, not run on hardware
//
#include <hip/hip_runtime.h>
#include <hip/hip_bf16.h>

typedef __attribute__((ext_vector_type(16))) _Float16 v16h;
typedef __attribute__((ext_vector_type(8)))  float    v8f;
typedef __attribute__((ext_vector_type(4))) unsigned int u32x4;
typedef __attribute__((ext_vector_type(8))) int          i32x8;
typedef __attribute__((ext_vector_type(4))) int          i32x4;

#define BATCH 128
#define QLEN  30
#define DLEN  256
#define EMB   300
#define EMBP  320   // K padded to multiple of 32
#define COUT  128
#define NK    11
#define QROWS 36    // 32 tile rows + 2 conv-shift rows + align (rows >=30 zero)
#define DROWS 260   // 256 + 2 conv-shift rows + align (rows >=256 zero)

#if defined(__has_builtin)
#if __has_builtin(__builtin_amdgcn_tensor_load_to_lds) && \
    __has_builtin(__builtin_amdgcn_s_wait_tensorcnt)
#define USE_TDM 1
#endif
#endif
#ifndef USE_TDM
#define USE_TDM 0
#endif

// RBF kernel constants: mu = [1.0, 0.9, 0.7, ..., -0.9]; sigma = [1e-3, 0.1 x10]
__constant__ float c_mu[NK]  = {1.0f, 0.9f, 0.7f, 0.5f, 0.3f, 0.1f,
                                -0.1f, -0.3f, -0.5f, -0.7f, -0.9f};
__constant__ float c_is2[NK] = {500000.0f, 50.0f, 50.0f, 50.0f, 50.0f, 50.0f,
                                50.0f, 50.0f, 50.0f, 50.0f, 50.0f};

// ---- WMMA fragment loaders (contiguous-K order, same for A and B, wave32) ----
// A 16x32 f16: lane<16 -> row=lane, K 0..15 contiguous; lane>=16 same rows, K 16..31.
// Caller passes p = &A[row*stride + kbase + 16*(lane>>4)]; chunks at p and p+8.
__device__ inline v16h load_a_frag(const _Float16* p) {
  struct alignas(32) T { float4 x, y; } t;
  t.x = *reinterpret_cast<const float4*>(p);
  t.y = *reinterpret_cast<const float4*>(p + 8);
  return *reinterpret_cast<v16h*>(&t);
}
// B 32x16 f16: lane<16 -> col=lane, K 0..15 contiguous; lane>=16 -> K 16..31.
// Caller passes p = &Bmem[col*stride + kbase + 16*(lane>>4)]; 32 contiguous bytes.
__device__ inline v16h load_b_frag(const _Float16* p) {
  struct alignas(32) T { float4 x, y; } t;
  t.x = *reinterpret_cast<const float4*>(p);
  t.y = *reinterpret_cast<const float4*>(p + 8);
  return *reinterpret_cast<v16h*>(&t);
}

#if USE_TDM
// ---- TDM descriptor for one 16x128-f16 tile of dn, padded into LDS ----
// Tensor: 2D, data_size=2B, dims (128, 256), stride0 = 128 elements.
// Tile: 128 x 16; LDS pad: 4 dwords after every 64 dwords (one 256B row) ->
// row pitch 272B, bank-conflict-free 16-lane reads.
#define TILE_PITCH_B 272
#define TILE_PITCH_H (TILE_PITCH_B / 2)
#define TILE_LDS_B   (16 * TILE_PITCH_B)

__device__ inline void tdm_load_tile(unsigned int lds_addr, unsigned long long gaddr) {
  unsigned int ga_lo = __builtin_amdgcn_readfirstlane((unsigned int)gaddr);
  unsigned int ga_hi = __builtin_amdgcn_readfirstlane((unsigned int)(gaddr >> 32));
  unsigned int la    = __builtin_amdgcn_readfirstlane(lds_addr);
  u32x4 g0;
  g0[0] = 1u;                                   // count=1, user load descriptor
  g0[1] = la;                                   // lds_addr (bytes)
  g0[2] = ga_lo;                                // global_addr[31:0]
  g0[3] = (ga_hi & 0x01FFFFFFu) | 0x80000000u;  // global_addr[56:32] | type=2
  i32x8 g1;
  g1[0] = (int)((1u << 16) |    // data_size = 2 bytes
                (1u << 20) |    // pad_enable
                (5u << 22) |    // pad_interval: 64 dwords
                (3u << 25));    // pad_amount: 4 dwords
  g1[1] = (int)((unsigned)COUT << 16);  // tensor_dim0 = 128 (bits [63:48])
  g1[2] = (int)((unsigned)DLEN << 16);  // tensor_dim1 = 256 (bits [95:80])
  g1[3] = (int)((unsigned)COUT << 16);  // tile_dim0 = 128 (bits [127:112])
  g1[4] = 16;                           // tile_dim1 = 16 (bits [143:128])
  g1[5] = COUT;                         // tensor_dim0_stride = 128
  g1[6] = 0;
  g1[7] = 0;
  i32x4 g2; g2[0] = 1; g2[1] = 0; g2[2] = 0; g2[3] = 0;  // tensor_dim2=1 (unused)
  i32x4 g3; g3[0] = 0; g3[1] = 0; g3[2] = 0; g3[3] = 0;
  i32x8 g4;  // extension operand on the 6-arg toolchain variant: zero-filled
  g4[0] = 0; g4[1] = 0; g4[2] = 0; g4[3] = 0;
  g4[4] = 0; g4[5] = 0; g4[6] = 0; g4[7] = 0;
  __builtin_amdgcn_tensor_load_to_lds(g0, g1, g2, g3, g4, 0);
}
#endif

// ---- 1) embeddings f32 -> f16, padded [B][rowsPad][320], zeros in pad ----
__global__ void knrm_cvt_emb(const float* __restrict__ src, _Float16* __restrict__ dst,
                             int L, int rowsPad) {
  size_t idx = (size_t)blockIdx.x * blockDim.x + threadIdx.x;
  size_t total = (size_t)BATCH * rowsPad * EMBP;
  if (idx >= total) return;
  int col = (int)(idx % EMBP);
  size_t rg = idx / EMBP;
  int row = (int)(rg % rowsPad);
  int b = (int)(rg / rowsPad);
  float v = 0.f;
  if (row < L && col < EMB) v = src[((size_t)b * L + row) * EMB + col];
  dst[idx] = (_Float16)v;
}

// ---- 2) conv weights [C][300][g] f32 -> wtr [3][3][128][320] f16 (channel-major) ----
__global__ void knrm_cvt_w(const float* __restrict__ w1, const float* __restrict__ w2,
                           const float* __restrict__ w3, _Float16* __restrict__ dst) {
  size_t idx = (size_t)blockIdx.x * blockDim.x + threadIdx.x;
  size_t total = (size_t)9 * COUT * EMBP;
  if (idx >= total) return;
  int e = (int)(idx % EMBP);
  int c = (int)((idx / EMBP) % COUT);
  int j = (int)((idx / ((size_t)EMBP * COUT)) % 3);
  int g = (int)(idx / ((size_t)EMBP * COUT * 3));
  float v = 0.f;
  if (e < EMB && j <= g) {
    const float* w = (g == 0) ? w1 : (g == 1) ? w2 : w3;
    v = w[((size_t)c * EMB + e) * (g + 1) + j];
  }
  dst[idx] = (_Float16)v;
}

// ---- 3) n-gram conv as WMMA GEMM: out[b][g][row][c] = relu(sum_j A(row+j,:) . W_j(:,c) + bias) ----
__global__ __launch_bounds__(256) void knrm_conv(
    const _Float16* __restrict__ emb, int rowsPad,
    const _Float16* __restrict__ wtr,
    const float* __restrict__ bias1, const float* __restrict__ bias2,
    const float* __restrict__ bias3,
    float* __restrict__ out, int Mtiles) {
  const int b = blockIdx.x, g = blockIdx.y;
  const int wave = threadIdx.x >> 5, lane = threadIdx.x & 31;
  const int tile = blockIdx.z * 8 + wave;
  if (tile >= Mtiles * 8) return;
  const int mt = tile >> 3, nt = tile & 7;
  const _Float16* A0 = emb + (size_t)b * rowsPad * EMBP;
  const int arow = mt * 16 + (lane & 15);
  const int aoff = (lane >> 4) * 16;
  const int brow = nt * 16 + (lane & 15);
  const int boff = (lane >> 4) * 16;
  v8f acc = {0.f, 0.f, 0.f, 0.f, 0.f, 0.f, 0.f, 0.f};
  for (int j = 0; j <= g; ++j) {  // shifted GEMM per tap
    const _Float16* W  = wtr + ((size_t)(g * 3 + j) * COUT) * EMBP;
    const _Float16* pA = A0 + (size_t)(arow + j) * EMBP + aoff;
    const _Float16* pB = W + (size_t)brow * EMBP + boff;
#pragma unroll
    for (int kk = 0; kk < EMBP / 32; ++kk) {
      v16h a  = load_a_frag(pA + kk * 32);
      v16h bf = load_b_frag(pB + kk * 32);
      acc = __builtin_amdgcn_wmma_f32_16x16x32_f16(false, a, false, bf,
                                                   (short)0, acc, false, false);
    }
  }
  const float* bias = (g == 0) ? bias1 : (g == 1) ? bias2 : bias3;
  const float bv = bias[nt * 16 + (lane & 15)];
  const int Mrows = Mtiles * 16;
  float* O = out + ((size_t)(b * 3 + g) * Mrows) * COUT;
  const int col = nt * 16 + (lane & 15);
  const int rb = mt * 16 + ((lane >> 4) * 8);
#pragma unroll
  for (int r = 0; r < 8; ++r) {
    float v = acc[r] + bv;
    O[(size_t)(rb + r) * COUT + col] = v > 0.f ? v : 0.f;
  }
}

// ---- 4) L2-normalize rows f32 -> f16; zero fake pad rows ----
__global__ __launch_bounds__(128) void knrm_norm(const float* __restrict__ src,
                                                 _Float16* __restrict__ dst,
                                                 int rowsPerBG, int realRows) {
  __shared__ float red[128];
  const int row = blockIdx.x, t = threadIdx.x;
  const float x = src[(size_t)row * COUT + t];
  red[t] = x * x;
  __syncthreads();
  for (int s = 64; s > 0; s >>= 1) {
    if (t < s) red[t] += red[t + s];
    __syncthreads();
  }
  float scale = 0.f;
  if ((row % rowsPerBG) < realRows) scale = 1.f / (sqrtf(red[0]) + 1e-13f);
  dst[(size_t)row * COUT + t] = (_Float16)(x * scale);
}

// ---- 5) fused cosine-GEMM + RBF kernel pooling ----
// grid (B, 3 q-grams, 2 q M-tiles), block = 96 (3 waves, wave <-> d-gram t).
// B-tiles staged via TDM into LDS (double-buffered, TENSORcnt-tracked, padded
// pitch for conflict-free ds_load_b128), overlapped with exp/FMA VALU work.
__global__ __launch_bounds__(96) void knrm_pool(
    const _Float16* __restrict__ qn,   // [B][3][32][128] f16 (rows>=30 zero)
    const _Float16* __restrict__ dn,   // [B][3][256][128] f16
    const float* __restrict__ qmask,   // [B][30]
    const float* __restrict__ dmask,   // [B][256]
    float* __restrict__ pool2) {       // [B][9][11][2] per-Mtile partials
  const int b = blockIdx.x, qi = blockIdx.y, mt = blockIdx.z;
  const int t = threadIdx.x >> 5, lane = threadIdx.x & 31;
  const _Float16* A0 = qn + (size_t)(b * 3 + qi) * 32 * COUT;
  const _Float16* B0 = dn + (size_t)(b * 3 + t) * DLEN * COUT;
  const _Float16* pA = A0 + (size_t)(mt * 16 + (lane & 15)) * COUT + (lane >> 4) * 16;

#if USE_TDM
  __shared__ __align__(16) char ldsB[3][2][TILE_LDS_B];
#endif

  float acc[NK][8];
#pragma unroll
  for (int k = 0; k < NK; ++k)
#pragma unroll
    for (int r = 0; r < 8; ++r) acc[k][r] = 0.f;

  float mq[8];
  const int rb = mt * 16 + ((lane >> 4) * 8);
#pragma unroll
  for (int r = 0; r < 8; ++r) {
    int qr = rb + r;
    mq[r] = (qr < QLEN) ? qmask[b * QLEN + qr] : 0.f;
  }

#if USE_TDM
  const unsigned long long gbase = (unsigned long long)(uintptr_t)B0;
  tdm_load_tile((unsigned int)(uintptr_t)&ldsB[t][0][0], gbase);
#endif

  for (int nt = 0; nt < DLEN / 16; ++nt) {
#if USE_TDM
    const int buf = nt & 1;
    asm volatile("" ::: "memory");
    if (nt < DLEN / 16 - 1) {
      tdm_load_tile((unsigned int)(uintptr_t)&ldsB[t][(nt + 1) & 1][0],
                    gbase + (unsigned long long)(nt + 1) * (16 * COUT * 2));
      __builtin_amdgcn_s_wait_tensorcnt(1);
    } else {
      __builtin_amdgcn_s_wait_tensorcnt(0);
    }
    asm volatile("" ::: "memory");
    const _Float16* Bl = (const _Float16*)&ldsB[t][buf][0];
    const _Float16* pB = Bl + (size_t)(lane & 15) * TILE_PITCH_H + (lane >> 4) * 16;
#else
    const _Float16* pB = B0 + (size_t)(nt * 16 + (lane & 15)) * COUT + (lane >> 4) * 16;
#endif
    v8f c = {0.f, 0.f, 0.f, 0.f, 0.f, 0.f, 0.f, 0.f};
#pragma unroll
    for (int kk = 0; kk < COUT / 32; ++kk) {
      v16h a  = load_a_frag(pA + kk * 32);
      v16h bf = load_b_frag(pB + kk * 32);
      c = __builtin_amdgcn_wmma_f32_16x16x32_f16(false, a, false, bf,
                                                 (short)0, c, false, false);
    }
    const float md = dmask[b * DLEN + nt * 16 + (lane & 15)];
#pragma unroll
    for (int r = 0; r < 8; ++r) {
      float m = mq[r] * md;
      float cm = c[r] * m;  // masked cosine
#pragma unroll
      for (int k = 0; k < NK; ++k) {
        float d = cm - c_mu[k];
        acc[k][r] += __expf(-d * d * c_is2[k]) * m;
      }
    }
  }

  // Reduce over 16 columns (lanes within each 16-group share rows), then over row halves.
  float val[NK];
#pragma unroll
  for (int k = 0; k < NK; ++k) {
    float vk = 0.f;
#pragma unroll
    for (int r = 0; r < 8; ++r) {
      float s = acc[k][r];
      s += __shfl_xor(s, 1, 32);
      s += __shfl_xor(s, 2, 32);
      s += __shfl_xor(s, 4, 32);
      s += __shfl_xor(s, 8, 32);  // s = per_kernel_query[b, row, k] for d-gram t
      vk += mq[r] * 0.01f * __logf(fmaxf(s, 1e-10f));
    }
    vk += __shfl_xor(vk, 16, 32);  // combine rows 0-7 with 8-15 of this M-tile
    val[k] = vk;
  }
  if (lane == 0) {
    const int p = qi * 3 + t;
#pragma unroll
    for (int k = 0; k < NK; ++k)
      pool2[((size_t)(b * 9 + p) * NK + k) * 2 + mt] = val[k];
  }
}

// ---- 6) final dense: out[b] = sum_j pool[b][j] * dense_w[j] ----
__global__ __launch_bounds__(128) void knrm_dense(const float* __restrict__ pool2,
                                                  const float* __restrict__ dw,
                                                  float* __restrict__ out) {
  __shared__ float red[128];
  const int b = blockIdx.x, t = threadIdx.x;
  float v = 0.f;
  if (t < 99)
    v = (pool2[(size_t)(b * 99 + t) * 2] + pool2[(size_t)(b * 99 + t) * 2 + 1]) * dw[t];
  red[t] = v;
  __syncthreads();
  for (int s = 64; s > 0; s >>= 1) {
    if (t < s) red[t] += red[t + s];
    __syncthreads();
  }
  if (t == 0) out[b] = red[0];
}

extern "C" void kernel_launch(void* const* d_in, const int* in_sizes, int n_in,
                              void* d_out, int out_size, void* d_ws, size_t ws_size,
                              hipStream_t stream) {
  const float* qemb  = (const float*)d_in[0];
  const float* demb  = (const float*)d_in[1];
  const float* qmask = (const float*)d_in[2];
  const float* dmask = (const float*)d_in[3];
  const float* w1 = (const float*)d_in[4];
  const float* b1 = (const float*)d_in[5];
  const float* w2 = (const float*)d_in[6];
  const float* b2 = (const float*)d_in[7];
  const float* w3 = (const float*)d_in[8];
  const float* b3 = (const float*)d_in[9];
  const float* dw = (const float*)d_in[10];
  float* out = (float*)d_out;

  char* ws = (char*)d_ws;
  size_t off = 0;
  auto alloc = [&](size_t bytes) {
    char* p = ws + off;
    off += (bytes + 255) & ~(size_t)255;
    return p;
  };
  _Float16* embQ  = (_Float16*)alloc((size_t)BATCH * QROWS * EMBP * 2);
  _Float16* embD  = (_Float16*)alloc((size_t)BATCH * DROWS * EMBP * 2);
  _Float16* wtr   = (_Float16*)alloc((size_t)9 * COUT * EMBP * 2);
  float*    convQ = (float*)alloc((size_t)BATCH * 3 * 32 * COUT * 4);
  float*    convD = (float*)alloc((size_t)BATCH * 3 * DLEN * COUT * 4);
  _Float16* qn    = (_Float16*)alloc((size_t)BATCH * 3 * 32 * COUT * 2);
  _Float16* dn    = (_Float16*)alloc((size_t)BATCH * 3 * DLEN * COUT * 2);
  float*    pool2 = (float*)alloc((size_t)BATCH * 99 * 2 * 4);
  if (off > ws_size) return;  // insufficient scratch

  {  // 1) f16 conversion with padding
    size_t tq = (size_t)BATCH * QROWS * EMBP;
    knrm_cvt_emb<<<dim3((unsigned)((tq + 255) / 256)), 256, 0, stream>>>(qemb, embQ, QLEN, QROWS);
    size_t td = (size_t)BATCH * DROWS * EMBP;
    knrm_cvt_emb<<<dim3((unsigned)((td + 255) / 256)), 256, 0, stream>>>(demb, embD, DLEN, DROWS);
    size_t tw = (size_t)9 * COUT * EMBP;
    knrm_cvt_w<<<dim3((unsigned)((tw + 255) / 256)), 256, 0, stream>>>(w1, w2, w3, wtr);
  }
  // 2) convolutions (WMMA GEMMs)
  knrm_conv<<<dim3(BATCH, 3, 2), 256, 0, stream>>>(embQ, QROWS, wtr, b1, b2, b3, convQ, 2);
  knrm_conv<<<dim3(BATCH, 3, 16), 256, 0, stream>>>(embD, DROWS, wtr, b1, b2, b3, convD, 16);
  // 3) row normalization -> f16
  knrm_norm<<<dim3(BATCH * 3 * 32), 128, 0, stream>>>(convQ, qn, 32, QLEN);
  knrm_norm<<<dim3(BATCH * 3 * DLEN), 128, 0, stream>>>(convD, dn, DLEN, DLEN);
  // 4) fused cosine GEMM + RBF pooling (WMMA + TDM double-buffering)
  knrm_pool<<<dim3(BATCH, 3, 2), 96, 0, stream>>>(qn, dn, qmask, dmask, pool2);
  // 5) dense readout
  knrm_dense<<<dim3(BATCH), 128, 0, stream>>>(pool2, dw, out);
}
